// ScaleDotProductAttention_35167192219950
// MI455X (gfx1250) — matrix operand order repaired and checked by reading, not compiled, not run on hardware
//
#include <hip/hip_runtime.h>

// SDPA (unfused reference semantics: returns out AND full attn matrix).
// B=2, H=12, S=2048, D=64, fp32 in/out. WMMA f32<-f16 on the matrix pipes.

typedef __attribute__((ext_vector_type(16))) _Float16 v16h;
typedef __attribute__((ext_vector_type(8)))  float    v8f;

#define SEQ   2048
#define HD    64
#define QT    16          // q rows per workgroup
#define NWAVE 8
#define COLS_PER_ITER (NWAVE * 16)       // 128 keys per pass-1 iteration
#define P1_ITERS (SEQ / COLS_PER_ITER)   // 16

// LDS carve (dynamic shared):
//   sc  : [16][2048] f32  scores -> exp values        131072 B
//   pp  : [16][2048] f16  normalized probs             65536 B
//   stg : staging (Q 16x64 / K 128x64 / V 2x32x64) f16 16384 B
//   red : [16][17] f32 reduction pad                    1088 B
//   rowmax/rowsum : 16 f32 each                          128 B
#define SC_OFF   0
#define PP_OFF   (16 * SEQ * 4)
#define STG_OFF  (PP_OFF + 16 * SEQ * 2)
#define RED_OFF  (STG_OFF + 128 * HD * 2)
#define RMAX_OFF (RED_OFF + 16 * 17 * 4)
#define RSUM_OFF (RMAX_OFF + 16 * 4)
#define SMEM_BYTES (RSUM_OFF + 16 * 4)

__device__ __forceinline__ int a_frag_k(int e, int hi) {
    // 16-bit A-matrix 16x32 VGPR layout (ISA 7.12.2)
    int vg = e >> 1, r = e & 1;
    int kb = (vg < 4) ? (2 * vg) : (16 + 2 * (vg - 4));
    return kb + r + hi * 8;
}

__global__ __launch_bounds__(256)
void sdpa_fused_kernel(const float* __restrict__ qg,
                       const float* __restrict__ kg,
                       const float* __restrict__ vg,
                       float* __restrict__ outg,
                       float* __restrict__ attng)
{
    extern __shared__ char smem[];
    float*    sc   = (float*)(smem + SC_OFF);
    _Float16* pp   = (_Float16*)(smem + PP_OFF);
    _Float16* stg  = (_Float16*)(smem + STG_OFF);
    float*    red  = (float*)(smem + RED_OFF);
    float*    rmax = (float*)(smem + RMAX_OFF);
    float*    rsum = (float*)(smem + RSUM_OFF);

    const int tid  = threadIdx.x;
    const int lane = tid & 31;
    const int wave = tid >> 5;
    const int bh   = blockIdx.y;          // 0..23
    const int q0   = blockIdx.x * QT;     // q-tile base row

    const float* qbase = qg + ((size_t)bh * SEQ + q0) * HD;
    const float* kbase = kg + (size_t)bh * SEQ * HD;
    const float* vbase = vg + (size_t)bh * SEQ * HD;
    float* obase = outg  + ((size_t)bh * SEQ + q0) * HD;
    float* abase = attng + ((size_t)bh * SEQ + q0) * SEQ;

    const int M  = lane & 15;             // row within 16 (A / C-low half)
    const int N  = lane & 15;             // col within 16 (B / C)
    const int hi = lane >> 4;

    // ---- stage Q tile (16x64) as f16, build per-wave A fragments ----
    for (int i = tid; i < QT * HD; i += 256)
        stg[i] = (_Float16)qbase[i];
    __syncthreads();

    v16h aq0, aq1;
    #pragma unroll
    for (int e = 0; e < 16; ++e) {
        int K = a_frag_k(e, hi);
        aq0[e] = stg[M * HD + K];
        aq1[e] = stg[M * HD + 32 + K];
    }
    __syncthreads();

    // ---- pass 1: S_tile = (Q K^T) * 1/sqrt(D) into LDS score buffer ----
    for (int it = 0; it < P1_ITERS; ++it) {
        const int colbase = it * COLS_PER_ITER;
        for (int i = tid; i < COLS_PER_ITER * HD; i += 256)
            stg[i] = (_Float16)kbase[(size_t)colbase * HD + i];
        __syncthreads();

        // B fragment: B[k_dim][n] = K[key = wave*16 + n][d]  (32x16 f16 layout)
        const int krow = wave * 16 + N;
        v16h b0, b1;
        #pragma unroll
        for (int e = 0; e < 16; ++e) {
            int K = (e < 8) ? (hi * 8 + e) : (16 + hi * 8 + (e - 8));
            b0[e] = stg[krow * HD + K];
            b1[e] = stg[krow * HD + 32 + K];
        }

        v8f c = {};
        c = __builtin_amdgcn_wmma_f32_16x16x32_f16(false, aq0, false, b0,
                                                   (short)0, c, false, false);
        c = __builtin_amdgcn_wmma_f32_16x16x32_f16(false, aq1, false, b1,
                                                   (short)0, c, false, false);

        #pragma unroll
        for (int r = 0; r < 8; ++r) {
            int row = r + 8 * hi;
            sc[row * SEQ + colbase + wave * 16 + N] = c[r] * 0.125f; // 1/sqrt(64)
        }
        __syncthreads();
    }

    // ---- softmax over full rows (16 threads per row) ----
    {
        const int row = tid >> 4;   // 0..15
        const int sub = tid & 15;

        float m = -3.0e38f;
        for (int j = sub; j < SEQ; j += 16)
            m = fmaxf(m, sc[row * SEQ + j]);
        red[row * 17 + sub] = m;
        __syncthreads();
        if (sub == 0) {
            float mm = red[row * 17];
            #pragma unroll
            for (int t = 1; t < 16; ++t) mm = fmaxf(mm, red[row * 17 + t]);
            rmax[row] = mm;
        }
        __syncthreads();

        const float mm = rmax[row];
        float s = 0.0f;
        for (int j = sub; j < SEQ; j += 16) {
            float e = __expf(sc[row * SEQ + j] - mm);
            sc[row * SEQ + j] = e;
            s += e;
        }
        red[row * 17 + sub] = s;
        __syncthreads();
        if (sub == 0) {
            float ss = 0.0f;
            #pragma unroll
            for (int t = 0; t < 16; ++t) ss += red[row * 17 + t];
            rsum[row] = ss;
        }
        __syncthreads();

        const float inv = 1.0f / rsum[row];
        for (int j = sub; j < SEQ; j += 16) {
            float p = sc[row * SEQ + j] * inv;
            pp[row * SEQ + j] = (_Float16)p;          // f16 copy for pass 2
            abase[(size_t)row * SEQ + j] = p;         // attn output (single write)
        }
        __syncthreads();
    }

    // ---- pass 2: out = P @ V. 8 waves = 4 d-tiles x 2 K-halves ----
    const int tile  = wave & 3;          // d columns tile*16 .. tile*16+15
    const int khalf = wave >> 2;         // key range [khalf*1024, khalf*1024+1024)
    v8f co = {};

    for (int i = 0; i < 32; ++i) {       // 32 chunks of K=32 per half
        // stage V rows [i*32, i*32+32) of both halves: [2][32][64] f16
        for (int t = tid; t < 2 * 32 * HD; t += 256) {
            int half = t / (32 * HD);
            int off  = t % (32 * HD);
            stg[t] = (_Float16)vbase[((size_t)half * 1024 + i * 32) * HD + off];
        }
        __syncthreads();

        const int kcg = khalf * 1024 + i * 32;   // global key base for this wave
        v16h a, b;
        #pragma unroll
        for (int e = 0; e < 16; ++e) {
            int K = a_frag_k(e, hi);
            a[e] = pp[M * SEQ + kcg + K];
            int Kb = (e < 8) ? (hi * 8 + e) : (16 + hi * 8 + (e - 8));
            b[e] = stg[khalf * 32 * HD + Kb * HD + tile * 16 + N];
        }
        co = __builtin_amdgcn_wmma_f32_16x16x32_f16(false, a, false, b,
                                                    (short)0, co, false, false);
        __syncthreads();
    }

    // combine the two K-halves via LDS (sc region is free now)
    float* part = sc;   // [4 tiles][16][16]
    if (khalf == 1) {
        #pragma unroll
        for (int r = 0; r < 8; ++r)
            part[(tile * 16 + (r + 8 * hi)) * 16 + N] = co[r];
    }
    __syncthreads();
    if (khalf == 0) {
        #pragma unroll
        for (int r = 0; r < 8; ++r) {
            int row = r + 8 * hi;
            obase[(size_t)row * HD + tile * 16 + N] =
                co[r] + part[(tile * 16 + row) * 16 + N];
        }
    }
}

extern "C" void kernel_launch(void* const* d_in, const int* in_sizes, int n_in,
                              void* d_out, int out_size, void* d_ws, size_t ws_size,
                              hipStream_t stream) {
    const float* q = (const float*)d_in[0];
    const float* k = (const float*)d_in[1];
    const float* v = (const float*)d_in[2];

    const int B = 2, H = 12;
    // d_out = [out (B,H,S,D)] ++ [attn (B,H,S,S)], flat, in return order
    float* out  = (float*)d_out;
    float* attn = (float*)d_out + (size_t)B * H * SEQ * HD;

    // Raise dynamic-LDS cap (CDNA5 WGP has 320KB); host-side attr set is
    // deterministic and graph-capture safe (not a stream op).
    (void)hipFuncSetAttribute((const void*)sdpa_fused_kernel,
                              hipFuncAttributeMaxDynamicSharedMemorySize,
                              (int)SMEM_BYTES);

    dim3 grid(SEQ / QT, B * H);   // (128, 24)
    dim3 block(256);
    sdpa_fused_kernel<<<grid, block, SMEM_BYTES, stream>>>(q, k, v, out, attn);
}
